// ZSDPredictor_85830626443973
// MI455X (gfx1250) — compile-verified, matrix operand order checked by reading, not run on hardware
//
#include <hip/hip_runtime.h>
#include <hip/hip_bf16.h>
#include <math.h>

typedef __attribute__((ext_vector_type(16))) _Float16 v16h;
typedef __attribute__((ext_vector_type(8)))  float    v8f;
typedef __attribute__((ext_vector_type(4)))  int      v4i;

#define N_ROWS   65536
#define IC       1024
#define NCLS     81
#define EDIM     300
#define NBOX     (4*NCLS)   // 324

#define NF_PAD   304        // pv cols padded -> 19 tiles of 16
#define NB_PAD   336        // bbox cols padded -> 21 tiles of 16
#define NT_F     19
#define NT_B     21
#define NT_ALL   40         // 8 waves * 5 jobs
#define KS_PAD   320        // scores K (300) padded to 10 chunks of 32
#define NE_PAD   96         // scores N (81) padded -> 6 tiles of 16
#define BM       32         // rows per block (2 M-tiles)

#define NE_BYTES (NE_PAD*KS_PAD*2)   // 61440

// workspace byte offsets
#define WF16_OFF 0
#define WB16_OFF (WF16_OFF + NF_PAD*IC*2)          // 622592
#define NE16_OFF (WB16_OFF + NB_PAD*IC*2)          // 1310720 (16B aligned)
#define BF_OFF   (NE16_OFF + NE_BYTES)             // 1372160
#define BB_OFF   (BF_OFF + NF_PAD*4)               // 1373376
#define PE_OFF   (BB_OFF + NB_PAD*4)               // 1374720

#if defined(__has_builtin)
#if __has_builtin(__builtin_amdgcn_global_load_async_to_lds_b128)
#define HAVE_ASYNC_LDS 1
#endif
#endif

#define AS1 __attribute__((address_space(1)))
#define AS3 __attribute__((address_space(3)))

union FragH { v16h v; uint4 q[2]; };

// ---------------------------------------------------------------------------
// 1) Convert weights to f16 with zero padding; pad biases.
// ---------------------------------------------------------------------------
__global__ __launch_bounds__(256)
void prep_weights(const float* __restrict__ Wf, const float* __restrict__ Wb,
                  const float* __restrict__ bf, const float* __restrict__ bb,
                  _Float16* __restrict__ wf16, _Float16* __restrict__ wb16,
                  float* __restrict__ bfp, float* __restrict__ bbp) {
  int idx = blockIdx.x * blockDim.x + threadIdx.x;
  int stride = gridDim.x * blockDim.x;
  for (int i = idx; i < NF_PAD * IC; i += stride) {
    int n = i >> 10, k = i & (IC - 1);
    wf16[i] = (_Float16)((n < EDIM) ? Wf[n * IC + k] : 0.0f);
  }
  for (int i = idx; i < NB_PAD * IC; i += stride) {
    int n = i >> 10, k = i & (IC - 1);
    wb16[i] = (_Float16)((n < NBOX) ? Wb[n * IC + k] : 0.0f);
  }
  for (int i = idx; i < NF_PAD; i += stride) bfp[i] = (i < EDIM) ? bf[i] : 0.0f;
  for (int i = idx; i < NB_PAD; i += stride) bbp[i] = (i < NBOX) ? bb[i] : 0.0f;
}

// ---------------------------------------------------------------------------
// 2) Small embedding path + triplet loss (single block; tiny work).
// ---------------------------------------------------------------------------
__global__ __launch_bounds__(256)
void embed_kernel(const float* __restrict__ se, const float* __restrict__ We,
                  const float* __restrict__ be,
                  _Float16* __restrict__ ne16, float* __restrict__ pe_ws,
                  float* __restrict__ loss_out) {
  __shared__ float inv_pe[NCLS];
  __shared__ float se_inv[NCLS];
  __shared__ float hinge[NCLS];
  const int t = threadIdx.x;

  for (int o = t; o < NCLS * EDIM; o += blockDim.x) {
    int c = o / EDIM, e = o % EDIM;
    float s = be[e];
    const float* sr = se + c * EDIM;
    const float* wr = We + e * EDIM;
    for (int k = 0; k < EDIM; ++k) s = fmaf(sr[k], wr[k], s);
    pe_ws[o] = s;
  }
  __threadfence();
  __syncthreads();

  if (t < NCLS) {
    float s = 0.0f;
    for (int k = 0; k < EDIM; ++k) { float v = pe_ws[t * EDIM + k]; s = fmaf(v, v, s); }
    inv_pe[t] = 1.0f / fmaxf(sqrtf(s), 1e-12f);
    float s2 = 0.0f;
    for (int k = 0; k < EDIM; ++k) { float v = se[t * EDIM + k]; s2 = fmaf(v, v, s2); }
    se_inv[t] = 1.0f / fmaxf(sqrtf(s2), 1e-12f);
  }
  __syncthreads();

  for (int o = t; o < NE_PAD * KS_PAD; o += blockDim.x) {
    int n = o / KS_PAD, k = o % KS_PAD;
    float v = (n < NCLS && k < EDIM) ? pe_ws[n * EDIM + k] * inv_pe[n] : 0.0f;
    ne16[o] = (_Float16)v;
  }

  if (t < NCLS) {
    const int i = t;
    float Sm[NCLS];
    for (int j = 0; j < NCLS; ++j) {
      float d = 0.0f;
      const float* a = se + i * EDIM;
      const float* b = se + j * EDIM;
      for (int k = 0; k < EDIM; ++k) d = fmaf(a[k], b[k], d);
      Sm[j] = d * se_inv[i] * se_inv[j];
    }
    float mx = -3.0e38f;
    for (int j = 0; j < NCLS; ++j)
      if (Sm[j] != 1.0f) mx = fmaxf(mx, Sm[j]);
    float den = 0.0f;
    for (int j = 0; j < NCLS; ++j)
      if (Sm[j] != 1.0f) den += __expf(Sm[j] - mx);
    den = fmaxf(den, 1e-30f);
    for (int j = 0; j < NCLS; ++j)
      if (Sm[j] != 1.0f) Sm[j] = __expf(Sm[j] - mx) / den;
    float b1 = -3.0e38f, b2 = -3.0e38f, mn = 3.0e38f;
    int i1 = 0, ms = 0, ls = 0;
    for (int j = 0; j < NCLS; ++j) {
      float v = Sm[j];
      if (v > b1) { b2 = b1; ms = i1; b1 = v; i1 = j; }
      else if (v > b2) { b2 = v; ms = j; }
      if (v <= mn) { mn = v; ls = j; }
    }
    float margin = b2 - mn;
    float dsim = 0.0f, ddis = 0.0f;
    for (int k = 0; k < EDIM; ++k) {
      float a = pe_ws[i * EDIM + k];
      float u = a - pe_ws[ms * EDIM + k];
      float w = a - pe_ws[ls * EDIM + k];
      dsim = fmaf(u, u, dsim);
      ddis = fmaf(w, w, ddis);
    }
    hinge[i] = fmaxf(0.0f, sqrtf(dsim) - sqrtf(ddis) + margin);
  }
  __syncthreads();
  if (t == 0) {
    float s = 0.0f;
    for (int j = 0; j < NCLS; ++j) s += hinge[j];
    loss_out[0] = s / (float)NCLS;
  }
}

// ---------------------------------------------------------------------------
// 3) Fused main kernel: per block = 32 rows of x (2 M-tiles).
//    Phase 0: async-stage ne16 (scores B-matrix) into dynamic LDS, overlapped
//             with phase 1.
//    Phase 1: pv (19 tiles) + bbox (21 tiles) GEMMs, K=1024, f16 WMMA.
//             Each wave owns 5 N-tiles x 2 M-tiles: B loaded once, 2 WMMAs.
//    Phase 2: l2norm of pv rows via LDS, convert to f16 in LDS.
//    Phase 3: scores = nv @ ne.T (K=320 padded), all operands from LDS.
// ---------------------------------------------------------------------------
__global__ __launch_bounds__(256)
void fused_main(const float* __restrict__ x,
                const _Float16* __restrict__ wf16,
                const _Float16* __restrict__ wb16,
                const float* __restrict__ bfp,
                const float* __restrict__ bbp,
                const _Float16* __restrict__ ne16,
                float* __restrict__ scores,
                float* __restrict__ bbox) {
  __shared__ float pv[BM * NF_PAD];      // 38912 B
  __shared__ _Float16 nvh[BM * KS_PAD];  // 20480 B
  __shared__ float invn[BM];
  extern __shared__ char dyn_smem[];     // NE_BYTES for staged ne16
  _Float16* ne_lds = (_Float16*)dyn_smem;

  const int t    = threadIdx.x;
  const int wave = t >> 5;
  const int lane = t & 31;
  const int nsub = lane & 15;      // A-matrix M / B-matrix N sub-index
  const int hi   = lane >> 4;      // lane half selects K sub-range
  const int row0 = blockIdx.x * BM;

  // ---- Phase 0: kick off async staging of ne16 into LDS ------------------
#if HAVE_ASYNC_LDS
  {
    const char* g = (const char*)ne16;
    for (int u = t; u < NE_BYTES / 16; u += 256) {
      __builtin_amdgcn_global_load_async_to_lds_b128(
          (AS1 v4i*)(g + (size_t)u * 16),
          (AS3 v4i*)(dyn_smem + (size_t)u * 16), 0, 0);
    }
  }
#else
  for (int u = t; u < NE_BYTES / 16; u += 256)
    ((uint4*)dyn_smem)[u] = ((const uint4*)ne16)[u];
#endif

  // ---- Phase 1: main GEMMs ------------------------------------------------
  const _Float16* wrow[5];
  int  ncol0[5];
  bool isF[5];
#pragma unroll
  for (int jj = 0; jj < 5; ++jj) {
    int gj = wave * 5 + jj;
    bool f = gj < NT_F;
    int n0 = f ? (gj << 4) : ((gj - NT_F) << 4);
    isF[jj] = f;
    ncol0[jj] = n0;
    const _Float16* base = f ? wf16 : wb16;
    wrow[jj] = base + (size_t)(n0 + nsub) * IC;
  }

  v8f zero = {};
  v8f acc0[5], acc1[5];
#pragma unroll
  for (int jj = 0; jj < 5; ++jj) { acc0[jj] = zero; acc1[jj] = zero; }

  const float* xrow0 = x + (size_t)(row0 + nsub) * IC;
  const float* xrow1 = x + (size_t)(row0 + 16 + nsub) * IC;

  for (int k0 = 0; k0 < IC; k0 += 32) {
    const int s0 = k0 + hi * 8;
    const int s1 = k0 + 16 + hi * 8;
    __builtin_prefetch((const void*)(xrow0 + k0 + 64), 0, 0);
    __builtin_prefetch((const void*)(xrow1 + k0 + 64), 0, 0);
    // A fragments for the two M-tiles: 16 f32 each, converted to f16
    float4 a0 = *(const float4*)(const void*)(xrow0 + s0);
    float4 a1 = *(const float4*)(const void*)(xrow0 + s0 + 4);
    float4 a2 = *(const float4*)(const void*)(xrow0 + s1);
    float4 a3 = *(const float4*)(const void*)(xrow0 + s1 + 4);
    float4 c0 = *(const float4*)(const void*)(xrow1 + s0);
    float4 c1 = *(const float4*)(const void*)(xrow1 + s0 + 4);
    float4 c2 = *(const float4*)(const void*)(xrow1 + s1);
    float4 c3 = *(const float4*)(const void*)(xrow1 + s1 + 4);
    FragH A0, A1;
    float af[16] = {a0.x, a0.y, a0.z, a0.w, a1.x, a1.y, a1.z, a1.w,
                    a2.x, a2.y, a2.z, a2.w, a3.x, a3.y, a3.z, a3.w};
    float cf[16] = {c0.x, c0.y, c0.z, c0.w, c1.x, c1.y, c1.z, c1.w,
                    c2.x, c2.y, c2.z, c2.w, c3.x, c3.y, c3.z, c3.w};
#pragma unroll
    for (int i = 0; i < 16; ++i) { A0.v[i] = (_Float16)af[i]; A1.v[i] = (_Float16)cf[i]; }
#pragma unroll
    for (int jj = 0; jj < 5; ++jj) {
      FragH B;
      B.q[0] = *(const uint4*)(const void*)(wrow[jj] + s0);
      B.q[1] = *(const uint4*)(const void*)(wrow[jj] + s1);
      acc0[jj] = __builtin_amdgcn_wmma_f32_16x16x32_f16(
          false, A0.v, false, B.v, (short)0, acc0[jj], false, false);
      acc1[jj] = __builtin_amdgcn_wmma_f32_16x16x32_f16(
          false, A1.v, false, B.v, (short)0, acc1[jj], false, false);
    }
  }

  // bias add + route: pv tiles -> LDS, bbox tiles -> global
#pragma unroll
  for (int jj = 0; jj < 5; ++jj) {
    const int col = ncol0[jj] + nsub;
    const float b = isF[jj] ? bfp[col] : bbp[col];
#pragma unroll
    for (int mt = 0; mt < 2; ++mt) {
#pragma unroll
      for (int r = 0; r < 8; ++r) {
        const float v = (mt ? acc1[jj][r] : acc0[jj][r]) + b;
        const int m = mt * 16 + r + hi * 8;
        if (isF[jj]) {
          pv[m * NF_PAD + col] = v;
        } else if (col < NBOX) {
          bbox[(size_t)(row0 + m) * NBOX + col] = v;
        }
      }
    }
  }
  __syncthreads();

  // ---- Phase 2: row norms + f16 normalized rows in LDS --------------------
  if (t < BM) {
    float s = 0.0f;
    for (int k = 0; k < EDIM; ++k) { float v = pv[t * NF_PAD + k]; s = fmaf(v, v, s); }
    invn[t] = 1.0f / fmaxf(sqrtf(s), 1e-12f);
  }
  __syncthreads();
  for (int o = t; o < BM * KS_PAD; o += 256) {
    int m = o / KS_PAD, k = o % KS_PAD;
    float v = (k < EDIM) ? pv[m * NF_PAD + k] * invn[m] : 0.0f;
    nvh[o] = (_Float16)v;
  }

  // drain async staging, then make everything visible to all waves
#if HAVE_ASYNC_LDS
#if __has_builtin(__builtin_amdgcn_s_wait_asynccnt)
  __builtin_amdgcn_s_wait_asynccnt(0);
#else
  asm volatile("s_wait_asynccnt 0x0" ::: "memory");
#endif
#endif
  __syncthreads();

  // ---- Phase 3: scores GEMM, 12 jobs (2 M-tiles x 6 N-tiles) --------------
  for (int sj = wave; sj < 12; sj += 8) {
    const int mt = sj / 6;
    const int nt = sj % 6;
    v8f c = zero;
    const int arow = (mt * 16 + nsub) * KS_PAD;
    const int brow = ((nt << 4) + nsub) * KS_PAD;
    for (int k0 = 0; k0 < KS_PAD; k0 += 32) {
      const int s0 = k0 + hi * 8;
      const int s1 = k0 + 16 + hi * 8;
      FragH A, B;
      A.q[0] = *(const uint4*)(const void*)&nvh[arow + s0];
      A.q[1] = *(const uint4*)(const void*)&nvh[arow + s1];
      B.q[0] = *(const uint4*)(const void*)&ne_lds[brow + s0];
      B.q[1] = *(const uint4*)(const void*)&ne_lds[brow + s1];
      c = __builtin_amdgcn_wmma_f32_16x16x32_f16(
          false, A.v, false, B.v, (short)0, c, false, false);
    }
    const int col = (nt << 4) + nsub;
    if (col < NCLS) {
#pragma unroll
      for (int r = 0; r < 8; ++r) {
        const int m = mt * 16 + r + hi * 8;
        scores[(size_t)(row0 + m) * NCLS + col] = c[r];
      }
    }
  }
}

// ---------------------------------------------------------------------------
extern "C" void kernel_launch(void* const* d_in, const int* in_sizes, int n_in,
                              void* d_out, int out_size, void* d_ws, size_t ws_size,
                              hipStream_t stream) {
  const float* x  = (const float*)d_in[0];
  const float* se = (const float*)d_in[1];
  const float* Wf = (const float*)d_in[2];
  const float* bf = (const float*)d_in[3];
  const float* We = (const float*)d_in[4];
  const float* be = (const float*)d_in[5];
  const float* Wb = (const float*)d_in[6];
  const float* bb = (const float*)d_in[7];

  char* ws = (char*)d_ws;
  _Float16* wf16 = (_Float16*)(ws + WF16_OFF);
  _Float16* wb16 = (_Float16*)(ws + WB16_OFF);
  _Float16* ne16 = (_Float16*)(ws + NE16_OFF);
  float*    bfp  = (float*)(ws + BF_OFF);
  float*    bbp  = (float*)(ws + BB_OFF);
  float*    pe   = (float*)(ws + PE_OFF);

  float* out    = (float*)d_out;
  float* scoresp = out;
  float* bboxp   = out + (size_t)N_ROWS * NCLS;
  float* lossp   = out + (size_t)N_ROWS * (NCLS + NBOX);

  prep_weights<<<512, 256, 0, stream>>>(Wf, Wb, bf, bb, wf16, wb16, bfp, bbp);
  embed_kernel<<<1, 256, 0, stream>>>(se, We, be, ne16, pe, lossp);
  fused_main<<<N_ROWS / BM, 256, NE_BYTES, stream>>>(x, wf16, wb16, bfp, bbp,
                                                     ne16, scoresp, bboxp);
}
